// LSTM_7352984010927
// MI455X (gfx1250) — matrix operand order repaired and checked by reading, not compiled, not run on hardware
//
#include <hip/hip_runtime.h>
#include <stdint.h>

typedef __attribute__((ext_vector_type(16))) _Float16 v16h;
typedef __attribute__((ext_vector_type(8)))  float    v8f;

#define WMMA16(A, B, C) \
  __builtin_amdgcn_wmma_f32_16x16x32_f16(false, (A), false, (B), (short)0, (C), false, false)

// ---- problem dims (fixed by the reference) ----
constexpr int SEQ  = 16;
constexpr int PRED = 15;
constexpr int POSE = 34;   // input / output feature dim
constexpr int HD   = 64;   // hidden dim -> 4*HD = 256 gate columns = 16 ntiles
// pre-swizzled B-fragment storage (dwords): [kt][nt][lane][8]
constexpr int GATE_DW = 2 * 16 * 32 * 8;  // 8192 dwords = 32 KB per gate-weight matrix
constexpr int FC_DW   = 2 * 3  * 32 * 8;  // 1536 dwords
constexpr int WS_WIH_E = 0;
constexpr int WS_WHH_E = GATE_DW;
constexpr int WS_WIH_D = 2 * GATE_DW;
constexpr int WS_WHH_D = 3 * GATE_DW;
constexpr int WS_FC    = 4 * GATE_DW;
constexpr int WS_TOTAL = 4 * GATE_DW + FC_DW;

// ---------------------------------------------------------------------------
// Prep kernel: convert f32 weights into f16 WMMA B-fragments (K x N order,
// per-lane layout from CDNA5 ISA: lane = {K-half, N}, dword j packs
// K = 8*half + 2j + p (+8 if j >= 4) for N = lane%16).
// ---------------------------------------------------------------------------
__global__ void lstm_swizzle_weights(const float* __restrict__ wih_e,
                                     const float* __restrict__ whh_e,
                                     const float* __restrict__ wih_d,
                                     const float* __restrict__ whh_d,
                                     const float* __restrict__ fcw,
                                     uint32_t* __restrict__ ws) {
  int tid = blockIdx.x * blockDim.x + threadIdx.x;
  if (tid >= WS_TOTAL) return;

  int mat, local;
  if      (tid < GATE_DW)     { mat = 0; local = tid; }
  else if (tid < 2 * GATE_DW) { mat = 1; local = tid - GATE_DW; }
  else if (tid < 3 * GATE_DW) { mat = 2; local = tid - 2 * GATE_DW; }
  else if (tid < 4 * GATE_DW) { mat = 3; local = tid - 3 * GATE_DW; }
  else                        { mat = 4; local = tid - 4 * GATE_DW; }

  const int j    = local & 7;
  const int lane = (local >> 3) & 31;
  const int hlf  = lane >> 4;
  const int n    = lane & 15;
  int nt, kt;
  if (mat < 4) { int g = local >> 8; nt = g & 15; kt = g >> 4; }
  else         { int g = local >> 8; nt = g % 3;  kt = g / 3; }

  union { uint32_t u; _Float16 h[2]; } d;
  d.u = 0;
#pragma unroll
  for (int p = 0; p < 2; ++p) {
    const int K  = 8 * hlf + 2 * j + p + (j >= 4 ? 8 : 0);     // K within 32-wide ktile
    const int kG = kt * 32 + K;
    const int nG = nt * 16 + n;
    float v = 0.0f;
    switch (mat) {
      case 0: if (kG < POSE) v = wih_e[nG * POSE + kG]; break;  // [256,34]
      case 1: v = whh_e[nG * HD + kG]; break;                   // [256,64]
      case 2: if (kG < POSE) v = wih_d[nG * POSE + kG]; break;
      case 3: v = whh_d[nG * HD + kG]; break;
      case 4: if (nG < POSE) v = fcw[nG * HD + kG]; break;      // fc_w [34,64], N-padded
    }
    d.h[p] = (_Float16)v;
  }
  ws[tid] = d.u;
}

// ---------------------------------------------------------------------------
// Async global -> LDS copy (CDNA5 GLOBAL_LOAD_ASYNC_TO_LDS_B128, ASYNCcnt)
// ---------------------------------------------------------------------------
__device__ __forceinline__ uint32_t lds_addr_of(const void* p) {
  return (uint32_t)(uintptr_t)(const __attribute__((address_space(3))) void*)p;
}

__device__ __forceinline__ void async_cp16(void* lds_dst, const void* gsrc) {
  asm volatile("global_load_async_to_lds_b128 %0, %1, off"
               :: "v"(lds_addr_of(lds_dst)), "v"((uint64_t)(uintptr_t)gsrc)
               : "memory");
}

__device__ __forceinline__ void wait_async0() {
#if __has_builtin(__builtin_amdgcn_s_wait_asynccnt)
  __builtin_amdgcn_s_wait_asynccnt(0);
#else
  asm volatile("s_wait_asynccnt 0x0" ::: "memory");
#endif
}

// ---------------------------------------------------------------------------
// Fragment loaders (LDS, layouts per ISA tables)
// ---------------------------------------------------------------------------
__device__ __forceinline__ v16h load_a_frag(const _Float16* buf, int r, int kt, int hlf) {
  // A 16x32 f16: lane holds row r=lane%16; dwords 0..3 = K[8*hlf .. +8),
  // dwords 4..7 = K[16+8*hlf .. +8) -> two contiguous 16B chunks.
  union { v16h v; uint4 q[2]; } A;
  const _Float16* p = buf + r * 64 + kt * 32 + 8 * hlf;
  A.q[0] = *reinterpret_cast<const uint4*>(p);
  A.q[1] = *reinterpret_cast<const uint4*>(p + 16);
  return A.v;
}

__device__ __forceinline__ v16h load_b_frag(const uint32_t* w, int kt, int nt, int NT, int lane) {
  union { v16h v; uint4 q[2]; } B;
  const uint32_t* p = w + (((kt * NT) + nt) * 32 + lane) * 8;
  B.q[0] = *reinterpret_cast<const uint4*>(p);
  B.q[1] = *reinterpret_cast<const uint4*>(p + 4);
  return B.v;
}

__device__ __forceinline__ v8f splat8(float x) {
  v8f r;
#pragma unroll
  for (int i = 0; i < 8; ++i) r[i] = x;
  return r;
}

// gfx1250 has native V_TANH_F32 -> 1 transcendental per activation.
// sigmoid(x) == 0.5*tanh(x/2) + 0.5 (exact identity).
__device__ __forceinline__ float ftanh(float x) {
#if __has_builtin(__builtin_amdgcn_tanhf)
  return __builtin_amdgcn_tanhf(x);
#else
  float e = __expf(-2.0f * x);
  return (1.0f - e) * __builtin_amdgcn_rcpf(1.0f + e);
#endif
}
__device__ __forceinline__ float fsig(float x) {
#if __has_builtin(__builtin_amdgcn_tanhf)
  return __builtin_amdgcn_tanhf(x * 0.5f) * 0.5f + 0.5f;
#else
  return __builtin_amdgcn_rcpf(1.0f + __expf(-x));
#endif
}

// One LSTM cell step for a wave's 16 batch rows.
// xbuf/hcur: [16][64] f16 A-slabs in LDS. hnxt: output h slab.
// cst: 32 per-lane cell-state f32 regs. bv[nt] = bias for column 16*nt + lane%16.
__device__ __forceinline__ void cell_step(const _Float16* xbuf, const _Float16* hcur,
                                          _Float16* hnxt, const uint32_t* wih,
                                          const uint32_t* whh, const float* bv,
                                          float* cst, int lane) {
  const int hlf = lane >> 4;
  const int n   = lane & 15;
  const v16h ax0 = load_a_frag(xbuf, n, 0, hlf);
  const v16h ax1 = load_a_frag(xbuf, n, 1, hlf);
  const v16h ah0 = load_a_frag(hcur, n, 0, hlf);
  const v16h ah1 = load_a_frag(hcur, n, 1, hlf);

#pragma unroll
  for (int tg = 0; tg < 4; ++tg) {           // gate N-subtile: unit u = 16*tg + n
    v8f acc[4];                               // i, f, g, o accumulators
#pragma unroll
    for (int g = 0; g < 4; ++g) {
      const int nt = g * 4 + tg;
      acc[g] = splat8(bv[nt]);
      acc[g] = WMMA16(ax0, load_b_frag(wih, 0, nt, 16, lane), acc[g]);
      acc[g] = WMMA16(ax1, load_b_frag(wih, 1, nt, 16, lane), acc[g]);
      acc[g] = WMMA16(ah0, load_b_frag(whh, 0, nt, 16, lane), acc[g]);
      acc[g] = WMMA16(ah1, load_b_frag(whh, 1, nt, 16, lane), acc[g]);
    }
#pragma unroll
    for (int j = 0; j < 8; ++j) {            // row m = 8*hlf + j, unit u = 16*tg + n
      const int idx = tg * 8 + j;
      const float iv = fsig(acc[0][j]);
      const float fv = fsig(acc[1][j]);
      const float gv = ftanh(acc[2][j]);
      const float ov = fsig(acc[3][j]);
      const float cn = fv * cst[idx] + iv * gv;
      cst[idx] = cn;
      hnxt[(hlf * 8 + j) * 64 + (tg * 16 + n)] = (_Float16)(ov * ftanh(cn));
    }
  }
}

// ---------------------------------------------------------------------------
// Main kernel: 256 threads = 8 waves; each wave owns 16 batch rows for all
// 16 encoder + 15 decoder steps. Weights live in LDS (re-staged enc->dec via
// async global->LDS DMA).
// ---------------------------------------------------------------------------
__global__ void __launch_bounds__(256)
lstm_seq2seq_kernel(const float* __restrict__ obs,
                    const uint32_t* __restrict__ wfrag,
                    const float* __restrict__ b_enc,
                    const float* __restrict__ b_dec,
                    const float* __restrict__ fc_b,
                    float* __restrict__ out, int batch) {
  __shared__ uint32_t s_wih[GATE_DW];
  __shared__ uint32_t s_whh[GATE_DW];
  __shared__ uint32_t s_wfc[FC_DW];
  __shared__ __align__(16) _Float16 s_x[8][16 * 64];
  __shared__ __align__(16) _Float16 s_h[8][2][16 * 64];

  const int tid  = threadIdx.x;
  const int wave = tid >> 5;
  const int lane = tid & 31;
  const int hlf  = lane >> 4;
  const int n    = lane & 15;
  const int B0   = blockIdx.x * 128 + wave * 16;   // first batch row of this wave

  // kick off async staging of encoder weight fragments into LDS (overlaps
  // with the zero-init / bias loads below; tracked by ASYNCcnt)
  {
    uint4*       dw = reinterpret_cast<uint4*>(s_wih);
    uint4*       dh = reinterpret_cast<uint4*>(s_whh);
    const uint4* sw = reinterpret_cast<const uint4*>(wfrag + WS_WIH_E);
    const uint4* sh = reinterpret_cast<const uint4*>(wfrag + WS_WHH_E);
    for (int i = tid; i < GATE_DW / 4; i += 256) {
      async_cp16(dw + i, sw + i);
      async_cp16(dh + i, sh + i);
    }
  }

  // zero per-wave x slab (pad cols 34..63 stay zero) and h ping-pong slabs
  for (int i = lane; i < 16 * 64; i += 32) {
    s_x[wave][i]    = (_Float16)0.0f;
    s_h[wave][0][i] = (_Float16)0.0f;
    s_h[wave][1][i] = (_Float16)0.0f;
  }

  float cst[32];
#pragma unroll
  for (int i = 0; i < 32; ++i) cst[i] = 0.0f;

  float bv[16];
#pragma unroll
  for (int nt = 0; nt < 16; ++nt) bv[nt] = b_enc[nt * 16 + n];

  wait_async0();
  __syncthreads();

  int cur = 0;
  // ---------------- encoder ----------------
#pragma unroll 1
  for (int t = 0; t < SEQ; ++t) {
    // stage obs_s[t] tile -> f16 x slab (coalesced-ish, 544 scalars / wave)
    for (int idx = lane; idx < 16 * POSE; idx += 32) {
      const int m = idx / POSE, k = idx % POSE;
      s_x[wave][m * 64 + k] =
          (_Float16)obs[((size_t)t * batch + B0 + m) * POSE + k];
    }
    if (t + 1 < SEQ)  // prefetch next timestep's tile (global_prefetch_b8)
      __builtin_prefetch(&obs[((size_t)(t + 1) * batch + B0 + n) * POSE], 0, 0);
    __syncthreads();
    cell_step(s_x[wave], s_h[wave][cur], s_h[wave][cur ^ 1], s_wih, s_whh, bv, cst, lane);
    __syncthreads();
    cur ^= 1;
  }

  // ---------------- swap to decoder weights (async DMA again) ----------------
  __syncthreads();
  {
    uint4*       dw = reinterpret_cast<uint4*>(s_wih);
    uint4*       dh = reinterpret_cast<uint4*>(s_whh);
    uint4*       df = reinterpret_cast<uint4*>(s_wfc);
    const uint4* sw = reinterpret_cast<const uint4*>(wfrag + WS_WIH_D);
    const uint4* sh = reinterpret_cast<const uint4*>(wfrag + WS_WHH_D);
    const uint4* sf = reinterpret_cast<const uint4*>(wfrag + WS_FC);
    for (int i = tid; i < GATE_DW / 4; i += 256) {
      async_cp16(dw + i, sw + i);
      async_cp16(dh + i, sh + i);
    }
    for (int i = tid; i < FC_DW / 4; i += 256) async_cp16(df + i, sf + i);
  }
#pragma unroll
  for (int nt = 0; nt < 16; ++nt) bv[nt] = b_dec[nt * 16 + n];
  float fcb[3];
#pragma unroll
  for (int tt = 0; tt < 3; ++tt) {
    const int u = tt * 16 + n;
    fcb[tt] = (u < POSE) ? fc_b[u] : 0.0f;
  }
  wait_async0();
  __syncthreads();

  // ---------------- decoder (s_x currently holds obs_s[15] == "last") -------
#pragma unroll 1
  for (int st = 0; st < PRED; ++st) {
    _Float16* hnxt = s_h[wave][cur ^ 1];
    cell_step(s_x[wave], s_h[wave][cur], hnxt, s_wih, s_whh, bv, cst, lane);
    __syncthreads();
    // fc: curr = h @ fc_w.T + fc_b  (N padded 34->48 => 3 ntiles)
    const v16h ha0 = load_a_frag(hnxt, n, 0, hlf);
    const v16h ha1 = load_a_frag(hnxt, n, 1, hlf);
#pragma unroll
    for (int tt = 0; tt < 3; ++tt) {
      v8f acc = splat8(fcb[tt]);
      acc = WMMA16(ha0, load_b_frag(s_wfc, 0, tt, 3, lane), acc);
      acc = WMMA16(ha1, load_b_frag(s_wfc, 1, tt, 3, lane), acc);
      const int u = tt * 16 + n;
      if (u < POSE) {
#pragma unroll
        for (int j = 0; j < 8; ++j) {
          const int m = hlf * 8 + j;
          const float v = acc[j];
          out[((size_t)st * batch + B0 + m) * POSE + u] = v;  // pred_s[st][b][u]
          s_x[wave][m * 64 + u] = (_Float16)v;                // feedback input
        }
      }
    }
    __syncthreads();
    cur ^= 1;
  }
}

// ---------------------------------------------------------------------------
extern "C" void kernel_launch(void* const* d_in, const int* in_sizes, int n_in,
                              void* d_out, int out_size, void* d_ws, size_t ws_size,
                              hipStream_t stream) {
  const float* obs   = (const float*)d_in[0];
  const float* wih_e = (const float*)d_in[1];
  const float* whh_e = (const float*)d_in[2];
  const float* b_enc = (const float*)d_in[3];
  const float* wih_d = (const float*)d_in[4];
  const float* whh_d = (const float*)d_in[5];
  const float* b_dec = (const float*)d_in[6];
  const float* fcw   = (const float*)d_in[7];
  const float* fcb   = (const float*)d_in[8];
  float*       out   = (float*)d_out;
  uint32_t*    ws    = (uint32_t*)d_ws;

  const int batch = in_sizes[0] / (SEQ * POSE);   // 32768

  lstm_swizzle_weights<<<(WS_TOTAL + 255) / 256, 256, 0, stream>>>(
      wih_e, whh_e, wih_d, whh_d, fcw, ws);
  lstm_seq2seq_kernel<<<batch / 128, 256, 0, stream>>>(
      obs, ws, b_enc, b_dec, fcb, out, batch);
}
